// _PointnetSAModuleBase_81037442941231
// MI455X (gfx1250) — matrix operand order repaired and checked by reading, not compiled, not run on hardware
//
#include <hip/hip_runtime.h>

typedef __attribute__((ext_vector_type(16))) _Float16 v16h;
typedef __attribute__((ext_vector_type(8)))  float    v8f;

#define B_    8
#define N_    16384
#define C_    64
#define S_    1024
#define K_    32
#define R2_   0.01f     // radius^2
#define CIN0  96        // 3 + 64 = 67 padded to 96 (multiple of 32 for WMMA K)
#define FPS_T 256
#define PPT   (N_ / FPS_T)

// ---------------------------------------------------------------------------
// WMMA fragment helpers (layouts per CDNA5 ISA 7.12.2, wave32)
// ---------------------------------------------------------------------------
// A 16x32 f16: lane L holds row M=L%16; elems 0..7 = K=base..base+7,
// elems 8..15 = K=base+16..base+23, base = (L<16 ? 0 : 8).
__device__ inline v16h load_a16(const _Float16* base, int m0, int k0, int ld) {
  int lane = threadIdx.x & 31;
  const _Float16* p = base + (m0 + (lane & 15)) * ld + k0 + ((lane & 16) ? 8 : 0);
  v16h a;
#pragma unroll
  for (int j = 0; j < 8; ++j) { a[j] = p[j]; a[8 + j] = p[16 + j]; }
  return a;
}

// B 32x16 f16 (weights stored col-major [n][k]): lane L holds col N=L%16;
// elems 0..7 = K=base..base+7, elems 8..15 = K=base+16..base+23, base = (L<16 ? 0 : 8).
__device__ inline v16h load_b16(const _Float16* w, int k0, int n0, int ldk) {
  int lane = threadIdx.x & 31;
  const _Float16* p = w + (size_t)(n0 + (lane & 15)) * ldk + k0 + ((lane & 16) ? 8 : 0);
  v16h b;
#pragma unroll
  for (int j = 0; j < 8; ++j) { b[j] = p[j]; b[8 + j] = p[16 + j]; }
  return b;
}

// C/D 16x16 f32: VGPR v <-> row M = v + (lane>=16 ? 8 : 0), col N = lane%16.
__device__ inline void store_relu_h(v8f acc, _Float16* out, int m0, int n0, int ld,
                                    const float* __restrict__ bias) {
  int lane = threadIdx.x & 31;
  int n = lane & 15;
  int mh = (lane & 16) ? 8 : 0;
  float bv = bias[n0 + n];
#pragma unroll
  for (int v = 0; v < 8; ++v) {
    float x = acc[v] + bv;
    out[(m0 + mh + v) * ld + n0 + n] = (_Float16)(x > 0.f ? x : 0.f);
  }
}

__device__ inline void store_bias_f(v8f acc, float* out, int m0, int n0, int ld,
                                    const float* __restrict__ bias) {
  int lane = threadIdx.x & 31;
  int n = lane & 15;
  int mh = (lane & 16) ? 8 : 0;
  float bv = bias[n0 + n];
#pragma unroll
  for (int v = 0; v < 8; ++v) out[(m0 + mh + v) * ld + n0 + n] = acc[v] + bv;
}

// ---------------------------------------------------------------------------
// Kernel 1: furthest point sampling. One block per batch; thread t owns points
// t, t+256, ... (coalesced); min-dists live in registers; block argmax
// reduction with lowest-index tie-break (matches jnp.argmax).
// ---------------------------------------------------------------------------
__global__ __launch_bounds__(FPS_T) void fps_kernel(const float* __restrict__ xyz,
                                                    float* __restrict__ new_xyz,
                                                    int* __restrict__ inds) {
  int b = blockIdx.x, t = threadIdx.x;
  const float* X = xyz + (size_t)b * N_ * 3;
  __shared__ float s_best[FPS_T];
  __shared__ int   s_bidx[FPS_T];
  __shared__ float s_pt[3];

  float mind[PPT];
#pragma unroll
  for (int j = 0; j < PPT; ++j) mind[j] = 1e10f;

  if (t == 0) {
    inds[b * S_] = 0;
    float x0 = X[0], y0 = X[1], z0 = X[2];
    s_pt[0] = x0; s_pt[1] = y0; s_pt[2] = z0;
    new_xyz[(size_t)b * S_ * 3 + 0] = x0;
    new_xyz[(size_t)b * S_ * 3 + 1] = y0;
    new_xyz[(size_t)b * S_ * 3 + 2] = z0;
  }
  __syncthreads();

  for (int i = 1; i < S_; ++i) {
    float px = s_pt[0], py = s_pt[1], pz = s_pt[2];
    float best = -1.f; int bidx = 0;
    for (int j = 0; j < PPT; ++j) {
      int n = t + j * FPS_T;
      float dx = X[n * 3 + 0] - px;
      float dy = X[n * 3 + 1] - py;
      float dz = X[n * 3 + 2] - pz;
      float d = dx * dx + dy * dy + dz * dz;
      float md = mind[j];
      md = d < md ? d : md;
      mind[j] = md;
      if (md > best) { best = md; bidx = n; }  // strict > keeps lowest index
    }
    s_best[t] = best; s_bidx[t] = bidx;
    __syncthreads();
    for (int off = FPS_T / 2; off; off >>= 1) {
      if (t < off) {
        float ob = s_best[t + off]; int oi = s_bidx[t + off];
        if (ob > s_best[t] || (ob == s_best[t] && oi < s_bidx[t])) {
          s_best[t] = ob; s_bidx[t] = oi;
        }
      }
      __syncthreads();
    }
    if (t == 0) {
      int w = s_bidx[0];
      inds[b * S_ + i] = w;
      float wx = X[w * 3], wy = X[w * 3 + 1], wz = X[w * 3 + 2];
      s_pt[0] = wx; s_pt[1] = wy; s_pt[2] = wz;
      new_xyz[((size_t)b * S_ + i) * 3 + 0] = wx;
      new_xyz[((size_t)b * S_ + i) * 3 + 1] = wy;
      new_xyz[((size_t)b * S_ + i) * 3 + 2] = wz;
    }
    __syncthreads();
  }
}

// ---------------------------------------------------------------------------
// Kernel 2: convert weights f32 -> f16, pad cin 67->96, transpose to
// col-major [cout][cin] so WMMA B-fragments are two contiguous 16B loads per lane.
// ---------------------------------------------------------------------------
__global__ void wconv_kernel(const float* __restrict__ w0, const float* __restrict__ w1,
                             const float* __restrict__ w2,
                             _Float16* __restrict__ wt0, _Float16* __restrict__ wt1,
                             _Float16* __restrict__ wt2) {
  int t = blockIdx.x * blockDim.x + threadIdx.x;
  int stride = gridDim.x * blockDim.x;
  for (int i = t; i < 64 * CIN0; i += stride) {              // w0: (67,64) -> [64][96]
    int n = i / CIN0, k = i % CIN0;
    wt0[i] = (_Float16)(k < 67 ? w0[k * 64 + n] : 0.f);
  }
  for (int i = t; i < 64 * 64; i += stride) {                // w1: (64,64) -> [64][64]
    int n = i / 64, k = i % 64;
    wt1[i] = (_Float16)w1[k * 64 + n];
  }
  for (int i = t; i < 128 * 64; i += stride) {               // w2: (64,128) -> [128][64]
    int n = i / 64, k = i % 64;
    wt2[i] = (_Float16)w2[k * 128 + n];
  }
}

// ---------------------------------------------------------------------------
// Kernel 3 (fused): ball query (wave32 ballot) + gather + 3-layer WMMA MLP
// + maxpool. One block = 2 centers = 64 rows; all intermediates in LDS.
// ---------------------------------------------------------------------------
__global__ __launch_bounds__(256) void sa_fused_kernel(
    const float* __restrict__ xyz, const float* __restrict__ feat,
    const float* __restrict__ new_xyz,
    const _Float16* __restrict__ wt0, const _Float16* __restrict__ wt1,
    const _Float16* __restrict__ wt2,
    const float* __restrict__ b0, const float* __restrict__ b1,
    const float* __restrict__ b2, float* __restrict__ out_feat) {
  int t = threadIdx.x;
  int wave = t >> 5, lane = t & 31;
  int g = blockIdx.x;
  int b = g / (S_ / 2);
  int spair = g - b * (S_ / 2);
  const float* X = xyz + (size_t)b * N_ * 3;
  const float* F = feat + (size_t)b * N_ * C_;

  __shared__ _Float16 hA[64 * CIN0];   // 12 KB: layer-0 input / layer-2 input
  __shared__ _Float16 hB[64 * 64];     //  8 KB: layer-1 output
  __shared__ float    hC[64 * 128];    // 32 KB: layer-3 pre-relu output
  __shared__ int      nlist[2][K_];
  __shared__ int      s_cnt[2];

  // --- ball query: waves 0,1 each handle one center; ballot collects the
  // first K_ in-radius indices in ascending point order (PointNet++ pad rule)
  if (wave < 2) {
    int cg = b * S_ + spair * 2 + wave;
    float cx = new_xyz[cg * 3 + 0];
    float cy = new_xyz[cg * 3 + 1];
    float cz = new_xyz[cg * 3 + 2];
    int cnt = 0;
    for (int base = 0; base < N_ && cnt < K_; base += 32) {
      int n = base + lane;
      float dx = X[n * 3 + 0] - cx;
      float dy = X[n * 3 + 1] - cy;
      float dz = X[n * 3 + 2] - cz;
      bool in = (dx * dx + dy * dy + dz * dz) <= R2_;
      unsigned m = __builtin_amdgcn_ballot_w32(in);
      if (in) {
        int pos = cnt + __popc(m & ((1u << lane) - 1u));
        if (pos < K_) nlist[wave][pos] = n;
      }
      cnt += __popc(m);
    }
    if (lane == 0) {
      if (cnt == 0) { nlist[wave][0] = 0; cnt = 1; }
      s_cnt[wave] = cnt < K_ ? cnt : K_;
    }
  }
  __syncthreads();

  // --- gather into hA: row r = (center-local, k); cols 0..2 rel-xyz,
  // 3..66 features, 67..95 zero pad. 4 threads per row, 24 cols each.
  {
    int r = t >> 2;                 // 0..63
    int cq = t & 3;                 // column quarter
    int cl = r >> 5, k = r & 31;
    int cnt = s_cnt[cl];
    int kk = (k < cnt) ? k : 0;     // pad slot with first in-radius index
    int n = nlist[cl][kk];
    int cg = b * S_ + spair * 2 + cl;
    float cc0 = new_xyz[cg * 3 + 0], cc1 = new_xyz[cg * 3 + 1], cc2 = new_xyz[cg * 3 + 2];
    _Float16* dst = hA + r * CIN0 + cq * 24;
    for (int j = 0; j < 24; ++j) {
      int col = cq * 24 + j;
      float v;
      if (col == 0)       v = X[n * 3 + 0] - cc0;
      else if (col == 1)  v = X[n * 3 + 1] - cc1;
      else if (col == 2)  v = X[n * 3 + 2] - cc2;
      else if (col < 67)  v = F[(size_t)n * C_ + (col - 3)];
      else                v = 0.f;
      dst[j] = (_Float16)v;
    }
  }
  __syncthreads();

  // --- layer 1: hA(64x96) x wt0 -> hB(64x64), relu.  4x4 tiles, 2 per wave.
#pragma unroll
  for (int i = 0; i < 2; ++i) {
    int ti = wave * 2 + i, mt = ti >> 2, nt = ti & 3;
    v8f acc = {};
#pragma unroll
    for (int kb = 0; kb < 3; ++kb) {
      v16h a  = load_a16(hA, mt * 16, kb * 32, CIN0);
      v16h bf = load_b16(wt0, kb * 32, nt * 16, CIN0);
      acc = __builtin_amdgcn_wmma_f32_16x16x32_f16(false, a, false, bf,
                                                   (short)0, acc, false, false);
    }
    store_relu_h(acc, hB, mt * 16, nt * 16, 64, b0);
  }
  __syncthreads();

  // --- layer 2: hB(64x64) x wt1 -> hA reused as (64x64), relu.
#pragma unroll
  for (int i = 0; i < 2; ++i) {
    int ti = wave * 2 + i, mt = ti >> 2, nt = ti & 3;
    v8f acc = {};
#pragma unroll
    for (int kb = 0; kb < 2; ++kb) {
      v16h a  = load_a16(hB, mt * 16, kb * 32, 64);
      v16h bf = load_b16(wt1, kb * 32, nt * 16, 64);
      acc = __builtin_amdgcn_wmma_f32_16x16x32_f16(false, a, false, bf,
                                                   (short)0, acc, false, false);
    }
    store_relu_h(acc, hA, mt * 16, nt * 16, 64, b1);
  }
  __syncthreads();

  // --- layer 3: hA(64x64) x wt2 -> hC(64x128) f32 + bias. 4x8 tiles, 4/wave.
#pragma unroll
  for (int i = 0; i < 4; ++i) {
    int ti = wave * 4 + i, mt = ti >> 3, nt = ti & 7;
    v8f acc = {};
#pragma unroll
    for (int kb = 0; kb < 2; ++kb) {
      v16h a  = load_a16(hA, mt * 16, kb * 32, 64);
      v16h bf = load_b16(wt2, kb * 32, nt * 16, 64);
      acc = __builtin_amdgcn_wmma_f32_16x16x32_f16(false, a, false, bf,
                                                   (short)0, acc, false, false);
    }
    store_bias_f(acc, hC, mt * 16, nt * 16, 128, b2);
  }
  __syncthreads();

  // --- maxpool over K_ then relu (relu(max) == max(relu)); output (B,128,S)
  {
    int cl = t >> 7, ch = t & 127;
    float m = hC[(cl * 32) * 128 + ch];
    for (int k = 1; k < K_; ++k) {
      float v = hC[(cl * 32 + k) * 128 + ch];
      if (v > m) m = v;
    }
    if (m < 0.f) m = 0.f;
    int s = spair * 2 + cl;
    out_feat[((size_t)b * 128 + ch) * S_ + s] = m;
  }
}

// ---------------------------------------------------------------------------
extern "C" void kernel_launch(void* const* d_in, const int* in_sizes, int n_in,
                              void* d_out, int out_size, void* d_ws, size_t ws_size,
                              hipStream_t stream) {
  (void)in_sizes; (void)n_in; (void)out_size; (void)ws_size;
  const float* xyz  = (const float*)d_in[0];
  const float* feat = (const float*)d_in[1];
  const float* w0   = (const float*)d_in[2];
  const float* b0   = (const float*)d_in[3];
  const float* w1   = (const float*)d_in[4];
  const float* b1   = (const float*)d_in[5];
  const float* w2   = (const float*)d_in[6];
  const float* b2   = (const float*)d_in[7];

  float* out      = (float*)d_out;
  float* new_xyz  = out;                                       // B*S*3
  float* new_feat = out + (size_t)B_ * S_ * 3;                 // B*128*S
  int*   inds     = (int*)(out + (size_t)B_ * S_ * 3 + (size_t)B_ * 128 * S_);

  _Float16* wt0 = (_Float16*)d_ws;            // [64][96]
  _Float16* wt1 = wt0 + 64 * CIN0;            // [64][64]
  _Float16* wt2 = wt1 + 64 * 64;              // [128][64]

  hipLaunchKernelGGL(wconv_kernel, dim3(6), dim3(256), 0, stream,
                     w0, w1, w2, wt0, wt1, wt2);
  hipLaunchKernelGGL(fps_kernel, dim3(B_), dim3(FPS_T), 0, stream,
                     xyz, new_xyz, inds);
  hipLaunchKernelGGL(sa_fused_kernel, dim3(B_ * S_ / 2), dim3(256), 0, stream,
                     xyz, feat, new_xyz, wt0, wt1, wt2, b0, b1, b2, new_feat);
}
